// ResidualStack_6511170421760
// MI455X (gfx1250) — compile-verified, matrix operand order checked by reading, not run on hardware
//
#include <hip/hip_runtime.h>

typedef __attribute__((ext_vector_type(16))) __bf16 v16bf;
typedef __attribute__((ext_vector_type(8)))  __bf16 v8bf;
typedef __attribute__((ext_vector_type(8)))  float  v8f;

#define RC    128          // residual / skip channels
#define TMAX  8192         // fixed time stride for activation buffers
#define TILEN 64           // time steps per workgroup
#define PAD   136          // LDS row stride in bf16 (16B aligned, 272B rows)
#define NBLK  40
#define SKIPT 4096

static __device__ __forceinline__ unsigned short f32_to_bf16(float f) {
    union { float f; unsigned int u; } v; v.f = f;
    unsigned int u = v.u;
    u += 0x7FFFu + ((u >> 16) & 1u);   // round-to-nearest-even
    return (unsigned short)(u >> 16);
}

// tanh(y)*sigmoid(y), branch-free: u = e^{-y},
// tanh = (1-u^2)/(1+u^2), sigmoid = 1/(1+u).
// Clamp to +-20: saturation error < 2e-9, all intermediates finite.
static __device__ __forceinline__ float gatef(float y) {
    y = fminf(fmaxf(y, -20.0f), 20.0f);
    float u  = __expf(-y);
    float u2 = u * u;
    return (1.0f - u2) * __builtin_amdgcn_rcpf((1.0f + u2) * (1.0f + u));
}

// Prepped A-fragment layout (bf16):
// wq[((((blk*4+mat)*8+mt)*4+kt)*32+lane)*16 + e]
//   o = mt*16 + (lane&15)
//   k = kt*32 + (lane>>4)*8 + (e>>3)*16 + (e&7)
// mat: 0 = W_dil tap0, 1 = W_dil tap1, 2 = W_res, 3 = W_skip
__global__ __launch_bounds__(256) void prep_weights(
    const float* __restrict__ wd, const float* __restrict__ wr,
    const float* __restrict__ wsk, unsigned short* __restrict__ wq) {
    int i = blockIdx.x * 256 + threadIdx.x;
    const int total = NBLK * 4 * RC * RC;
    if (i >= total) return;
    int e    = i & 15;
    int lane = (i >> 4) & 31;
    int kt   = (i >> 9) & 3;
    int mt   = (i >> 11) & 7;
    int mat  = (i >> 14) & 3;
    int blk  = i >> 16;
    int o = mt * 16 + (lane & 15);
    int k = kt * 32 + (lane >> 4) * 8 + (e >> 3) * 16 + (e & 7);
    float v;
    if (mat == 0)      v = wd[(((size_t)blk * RC + o) * RC + k) * 2 + 0];
    else if (mat == 1) v = wd[(((size_t)blk * RC + o) * RC + k) * 2 + 1];
    else if (mat == 2) v = wr[((size_t)blk * RC + o) * RC + k];
    else               v = wsk[((size_t)blk * RC + o) * RC + k];
    wq[i] = f32_to_bf16(v);
}

// One-time: transpose input x [b][ch][t] f32 -> F0 [b][t][ch] f32 + H0 bf16
__global__ __launch_bounds__(256) void prep_act(
    const float* __restrict__ x, float* __restrict__ F0,
    unsigned short* __restrict__ H0) {
    int i = blockIdx.x * 256 + threadIdx.x;            // 4*128*8192 threads
    int t  = i & (TMAX - 1);
    int ch = (i >> 13) & (RC - 1);
    int b  = i >> 20;
    float v = x[((size_t)b * RC + ch) * TMAX + t];     // coalesced in t
    size_t o = ((size_t)b * TMAX + t) * RC + ch;
    F0[o] = v;
    H0[o] = f32_to_bf16(v);
}

static __device__ __forceinline__ v16bf load_afrag(
    const unsigned short* __restrict__ wq, int blk, int mat, int mt, int kt, int lane) {
    size_t off = ((((size_t)((blk * 4 + mat) * 8 + mt)) * 4 + kt) * 32 + lane) * 16;
    return *(const v16bf*)(wq + off);   // 32B coalesced per lane -> 2x global_load_b128
}

// tile: LDS [t][ch] bf16, row stride PAD. B matrix is K x N (channels x time):
// lane<16 -> column n=lane, K = {kb..kb+7, kb+16..kb+23}; lane>=16 mirrored (+8).
static __device__ __forceinline__ v16bf load_bfrag(
    const unsigned short* tile, int nt, int kt, int lane) {
    int t  = nt * 16 + (lane & 15);
    int kb = kt * 32 + (lane >> 4) * 8;
    const unsigned short* p = tile + t * PAD + kb;
    v8bf lo = *(const v8bf*)p;          // 16B aligned ds_load_b128
    v8bf hi = *(const v8bf*)(p + 16);
    return __builtin_shufflevector(lo, hi, 0,1,2,3,4,5,6,7,8,9,10,11,12,13,14,15);
}

__global__ __launch_bounds__(256) void wavenet_block(
    const float* __restrict__ xf_in,          // f32 [b][t][ch] residual stream
    const unsigned short* __restrict__ hf_in, // bf16 [b][t][ch] same values
    float* __restrict__ xf_out,
    unsigned short* __restrict__ hf_out,
    const unsigned short* __restrict__ wq,
    const float* __restrict__ b_res, const float* __restrict__ b_skip,
    float* __restrict__ skip_out,
    int blk, int dil, int Tin, int Tout, int skip_start) {

    __shared__ unsigned short XLT[TILEN * PAD];   // tap0 input, [t][ch] bf16
    __shared__ unsigned short XRT[TILEN * PAD];   // tap1 input (t+dil)
    __shared__ unsigned short GT [TILEN * PAD];   // gated activations

    const int tid  = threadIdx.x;
    const int lane = tid & 31;
    const int mt   = tid >> 5;                    // wave id -> M strip
    const int b    = blockIdx.y;
    const int t0   = blockIdx.x * TILEN;
    const int chb  = mt * 16 + (lane >> 4) * 8;   // C layout: vgpr r -> M = r (+8 hi lanes)

    // ---- hoist conv A-fragments (held in registers across all n-tiles) ----
    v16bf Adil[8];                                // [kt][tap]
    #pragma unroll
    for (int kt = 0; kt < 4; ++kt) {
        Adil[2 * kt + 0] = load_afrag(wq, blk, 0, mt, kt, lane);
        Adil[2 * kt + 1] = load_afrag(wq, blk, 1, mt, kt, lane);
    }

    // ---- stage bf16 rows into LDS (pure copy, no conversion) ----
    // 2048 chunks of 16B: {XLT,XRT} x 64 rows x 16 chunks/row
    #pragma unroll
    for (int it = 0; it < 8; ++it) {
        int i     = tid + it * 256;
        int which = i >> 10;                 // 0: XLT, 1: XRT
        int rr    = (i >> 4) & 63;
        int c8    = (i & 15) * 8;            // ushort offset within row
        int row   = t0 + rr + (which ? dil : 0);
        if (row > TMAX - 1) row = TMAX - 1;  // stay in-buffer; columns guarded later
        uint4 v = *(const uint4*)(hf_in + ((size_t)b * TMAX + row) * RC + c8);
        unsigned short* dst = (which ? XRT : XLT) + rr * PAD + c8;
        *(uint4*)dst = v;
    }
    __syncthreads();

    // ---- phase 1: dilated conv (two taps) + gating -> GT ----
    #pragma unroll
    for (int nt = 0; nt < 4; ++nt) {
        v8f acc = {};
        #pragma unroll
        for (int kt = 0; kt < 4; ++kt) {
            v16bf bl = load_bfrag(XLT, nt, kt, lane);
            acc = __builtin_amdgcn_wmma_f32_16x16x32_bf16(false, Adil[2 * kt + 0], false, bl,
                                                          (short)0, acc, false, false);
            v16bf br = load_bfrag(XRT, nt, kt, lane);
            acc = __builtin_amdgcn_wmma_f32_16x16x32_bf16(false, Adil[2 * kt + 1], false, br,
                                                          (short)0, acc, false, false);
        }
        int trow = nt * 16 + (lane & 15);
        #pragma unroll
        for (int r = 0; r < 8; r += 2) {
            unsigned int pk = (unsigned int)f32_to_bf16(gatef(acc[r]))
                            | ((unsigned int)f32_to_bf16(gatef(acc[r + 1])) << 16);
            *(unsigned int*)&GT[trow * PAD + chb + r] = pk;
        }
    }
    __syncthreads();

    // ---- phase 2: residual (mat=2) and skip (mat=3) 1x1 convs ----
    v16bf Ares[4], Askp[4];
    #pragma unroll
    for (int kt = 0; kt < 4; ++kt) {
        Ares[kt] = load_afrag(wq, blk, 2, mt, kt, lane);
        Askp[kt] = load_afrag(wq, blk, 3, mt, kt, lane);
    }
    float brv[8], bsv[8];
    *(float4*)&brv[0] = *(const float4*)(b_res  + blk * RC + chb);
    *(float4*)&brv[4] = *(const float4*)(b_res  + blk * RC + chb + 4);
    *(float4*)&bsv[0] = *(const float4*)(b_skip + blk * RC + chb);
    *(float4*)&bsv[4] = *(const float4*)(b_skip + blk * RC + chb + 4);

    #pragma unroll
    for (int nt = 0; nt < 4; ++nt) {
        v8f accR = {};
        v8f accS = {};
        #pragma unroll
        for (int kt = 0; kt < 4; ++kt) {
            v16bf g = load_bfrag(GT, nt, kt, lane);
            accR = __builtin_amdgcn_wmma_f32_16x16x32_bf16(false, Ares[kt], false, g,
                                                           (short)0, accR, false, false);
            accS = __builtin_amdgcn_wmma_f32_16x16x32_bf16(false, Askp[kt], false, g,
                                                           (short)0, accS, false, false);
        }
        int t = t0 + nt * 16 + (lane & 15);
        int s = t - skip_start;
        if (t < Tout) {
            // residual input: 8 consecutive channels of row t+dil -> 2x float4
            size_t rrow = ((size_t)b * TMAX + t + dil) * RC + chb;
            float rv[8];
            *(float4*)&rv[0] = *(const float4*)(xf_in + rrow);
            *(float4*)&rv[4] = *(const float4*)(xf_in + rrow + 4);
            float vo[8];
            #pragma unroll
            for (int r = 0; r < 8; ++r) vo[r] = accR[r] + brv[r] + rv[r];
            // f32 residual stream out: 2x float4
            size_t orow = ((size_t)b * TMAX + t) * RC + chb;
            *(float4*)(xf_out + orow)     = *(float4*)&vo[0];
            *(float4*)(xf_out + orow + 4) = *(float4*)&vo[4];
            // bf16 copy out (next block's GEMM input): 1x uint4
            unsigned int pk[4];
            #pragma unroll
            for (int j = 0; j < 4; ++j)
                pk[j] = (unsigned int)f32_to_bf16(vo[2 * j])
                      | ((unsigned int)f32_to_bf16(vo[2 * j + 1]) << 16);
            *(uint4*)(hf_out + orow) = *(uint4*)&pk[0];
            // skip output (layout fixed by reference: ch-major, s fastest)
            if (s >= 0) {
                #pragma unroll
                for (int r = 0; r < 8; ++r) {
                    skip_out[(((size_t)blk * 4 + b) * RC + chb + r) * SKIPT + s]
                        = accS[r] + bsv[r];
                }
            }
        }
    }
}

extern "C" void kernel_launch(void* const* d_in, const int* in_sizes, int n_in,
                              void* d_out, int out_size, void* d_ws, size_t ws_size,
                              hipStream_t stream) {
    const float* x   = (const float*)d_in[0];
    const float* wd  = (const float*)d_in[1];
    const float* wr  = (const float*)d_in[2];
    const float* br  = (const float*)d_in[3];
    const float* wsk = (const float*)d_in[4];
    const float* bsk = (const float*)d_in[5];
    // skip_size (d_in[6]) is a compile-time constant 4096 in the reference.

    const size_t WQ_BYTES = (size_t)NBLK * 4 * RC * RC * 2;    // 5,242,880
    const size_t ACT      = (size_t)4 * TMAX * RC;             // 4,194,304 elems
    char* p = (char*)d_ws;
    unsigned short* wq = (unsigned short*)p;        p += WQ_BYTES;
    float* F0 = (float*)p;                          p += ACT * 4;
    float* FA = (float*)p;                          p += ACT * 4;
    float* FB = (float*)p;                          p += ACT * 4;
    unsigned short* H0 = (unsigned short*)p;        p += ACT * 2;
    unsigned short* HA = (unsigned short*)p;        p += ACT * 2;
    unsigned short* HB = (unsigned short*)p;        p += ACT * 2;

    prep_weights<<<(int)((WQ_BYTES / 2 + 255) / 256), 256, 0, stream>>>(wd, wr, wsk, wq);
    prep_act<<<(int)((ACT + 255) / 256), 256, 0, stream>>>(x, F0, H0);

    int T = TMAX;
    const float* curF = F0; const unsigned short* curH = H0;
    float* nxtF = FA;       unsigned short* nxtH = HA;
    for (int i = 0; i < NBLK; ++i) {
        int dil  = 1 << (i % 10);
        int Tout = T - dil;
        dim3 grid((Tout + TILEN - 1) / TILEN, 4);
        wavenet_block<<<grid, 256, 0, stream>>>(
            curF, curH, nxtF, nxtH, wq, br, bsk, (float*)d_out,
            i, dil, T, Tout, Tout - SKIPT);
        curF = nxtF; curH = nxtH;
        if (nxtF == FA) { nxtF = FB; nxtH = HB; }
        else            { nxtF = FA; nxtH = HA; }
        T = Tout;
    }
}
